// trajectory2seq_15676630630574
// MI455X (gfx1250) — matrix operand order repaired and checked by reading, not compiled, not run on hardware
//
#include <hip/hip_runtime.h>
#include <hip/hip_bf16.h>

// Sizes from the reference
#define Hh   512
#define NG   2048   // 4*H
#define Bq   128
#define Tt   256
#define Vv   128
#define LWn  32

// Use gfx1250 async global->LDS copies for bf16 panel staging (inline asm).
#define USE_ASYNC_LDS 1

typedef __attribute__((ext_vector_type(16))) __bf16 v16bf;
typedef __attribute__((ext_vector_type(8)))  __bf16 v8bf;
typedef __attribute__((ext_vector_type(4)))  __bf16 v4bf;
typedef __attribute__((ext_vector_type(4)))  float  v4f;
typedef __attribute__((ext_vector_type(8)))  float  v8f;

// Native RNE f32->bf16 (backend picks the hardware cvt).
__device__ __forceinline__ unsigned short f2bfu(float f) {
  __bf16 b = (__bf16)f;
  unsigned short u; __builtin_memcpy(&u, &b, 2); return u;
}
__device__ __forceinline__ float bf2f(unsigned short u) {
  union { unsigned u; float f; } x; x.u = ((unsigned)u) << 16; return x.f;
}
__device__ __forceinline__ float sigm(float x) { return 1.0f / (1.0f + __expf(-x)); }

// B operand pre-packed fragment-linear: tile(kt,nt) -> 512 ushorts, lane-major,
// so each lane reads 16 contiguous bf16 (2 x b128 loads).
__device__ __forceinline__ v16bf load_b_packed(const unsigned short* Wp, int NT,
                                               int k0, int nt, int lane) {
  const unsigned short* p = Wp + (((size_t)(k0 >> 5) * NT + nt) << 9) + lane * 16;
  return *(const v16bf*)p;
}

// A fragment from an LDS panel [128 rows][32 k] of bf16 (stored as ushort).
__device__ __forceinline__ v16bf load_a_lds(const unsigned short* ldsA, int m0, int lane) {
  int half = lane >> 4, r = lane & 15;
  const unsigned short* row = ldsA + (m0 + r) * 32;
  v8bf lo = *(const v8bf*)(row + half * 16);
  v8bf hi = *(const v8bf*)(row + half * 16 + 8);
  return __builtin_shufflevector(lo, hi, 0,1,2,3,4,5,6,7,8,9,10,11,12,13,14,15);
}

// ---- panel staging helpers (fixed trip counts: no exec-mask guards) ----
// bf16 source: 128 rows x 32 cols = 512 x 16B chunks; optionally row-indexed (emb gather).
template <bool HasIdx>
__device__ __forceinline__ void stage_bf16_panel(const unsigned short* src, long ld, int k0,
                                                 unsigned short* ldsA, int tid, const int* idx) {
#pragma unroll
  for (int ii = 0; ii < 2; ++ii) {
    int c = tid + ii * 256;
    int row = c >> 2, part = c & 3;
    int rg = HasIdx ? idx[row] : row;
    const unsigned short* g = src + (long)rg * ld + k0 + part * 8;
#if USE_ASYNC_LDS
    unsigned ldst = (unsigned)(uintptr_t)ldsA + (unsigned)c * 16u;
    asm volatile("global_load_async_to_lds_b128 %0, %1, off"
                 :: "v"(ldst), "v"(g) : "memory");
#else
    *(uint4*)(ldsA + c * 8) = *(const uint4*)g;
#endif
  }
#if USE_ASYNC_LDS
  asm volatile("s_wait_asynccnt 0x0" ::: "memory");
#endif
}

// f32 source: b128 load -> 2x v_cvt_pk_bf16_f32 -> ds_store_b64 (vector convert).
__device__ __forceinline__ void stage_f32_panel(const float* src, long ld, int k0,
                                                unsigned short* ldsA, int tid) {
#pragma unroll
  for (int ii = 0; ii < 4; ++ii) {
    int c = tid + ii * 256;
    int row = c >> 3, part = c & 7;
    v4f f = *(const v4f*)(src + (long)row * ld + k0 + part * 4);
    v4bf h = __builtin_convertvector(f, v4bf);
    *(v4bf*)(ldsA + row * 32 + part * 4) = h;
  }
}

// ---------------- weight pre-pack / convert ----------------
__global__ __launch_bounds__(256) void pack_b_kernel(const float* W, int K, int N,
                                                     unsigned short* out) {
  long i = (long)blockIdx.x * 256 + threadIdx.x;
  if (i >= (long)K * N) return;
  int k = (int)(i / N), n = (int)(i % N);
  int half = (k >> 4) & 1;
  int j = k & 15;
  int lane = half * 16 + (n & 15);
  long tile = (long)(k >> 5) * (N >> 4) + (n >> 4);
  out[(tile << 9) + lane * 16 + j] = f2bfu(W[i]);
}

__global__ __launch_bounds__(256) void cvt_bf_kernel(const float* s, unsigned short* d, long n) {
  long i = (long)blockIdx.x * 256 + threadIdx.x;
  if (i < n) d[i] = f2bfu(s[i]);
}
__global__ __launch_bounds__(256) void zero_kernel(float* p, long n) {
  long i = (long)blockIdx.x * 256 + threadIdx.x;
  if (i < n) p[i] = 0.0f;
}
__global__ __launch_bounds__(256) void add_kernel(const float* a, const float* b, float* d, long n) {
  long i = (long)blockIdx.x * 256 + threadIdx.x;
  if (i < n) d[i] = a[i] + b[i];
}
__global__ __launch_bounds__(256) void copyf_kernel(const float* s, float* d, long n) {
  long i = (long)blockIdx.x * 256 + threadIdx.x;
  if (i < n) d[i] = s[i];
}
__global__ void set_tok_kernel(int* tok) {
  if (threadIdx.x < Bq) tok[threadIdx.x] = 1;   // SOS = 1
}

// ---------------- fused LSTM step (gates GEMM + activations) ----------------
// gates = in1 @ W1 + hprev @ W2 + bias (+ x2 @ Wih2 when MODE==0)
// MODE: 0 = no in1 (encoder l0, tiny K=2 x-part added scalar)
//       1 = bf16 in1 (encoder l1)     2 = bf16 in1, row-indexed (decoder emb[tok])
//       3 = f32 in1 (decoder layer 1)
// Block = one 16-col slice of H, all 4 gates, all 128 rows.  8 waves, 4 tiles/wave.
template <int MODE>
__global__ __launch_bounds__(256) void lstm_step_kernel(
    const unsigned short* in1_bf, const float* in1_f32, const int* in1_idx,
    long in1_ld, int K1, const unsigned short* W1p,
    const float* hprev, const unsigned short* W2p,
    const float* bias,
    const float* x2, long x2_ld, const float* Wih2,
    float* c_io, float* h_next,
    unsigned short* yout, long yout_ld, long yout_off)
{
  __shared__ unsigned short ldsA[Bq * 32];     // 8 KB bf16 A panel
  __shared__ float ldsG[4][Bq][16];            // 32 KB staged gates
  const int tid  = threadIdx.x;
  const int wave = tid >> 5, lane = tid & 31;
  const int jblk = blockIdx.x;                 // 0..31 (H/16)
  const int gate = wave >> 1;                  // one gate per wave pair
  const int rb0  = (wave & 1) * 4;             // row blocks rb0..rb0+3
  const int ntile = gate * (Hh >> 4) + jblk;   // column tile in [0,128)
  v8f acc[4] = {};

  // Part 1: sequence/embedding input
  if constexpr (MODE != 0) {
    for (int k0 = 0; k0 < K1; k0 += 32) {
      if constexpr (MODE == 1)      stage_bf16_panel<false>(in1_bf, in1_ld, k0, ldsA, tid, nullptr);
      else if constexpr (MODE == 2) stage_bf16_panel<true>(in1_bf, in1_ld, k0, ldsA, tid, in1_idx);
      else                          stage_f32_panel(in1_f32, in1_ld, k0, ldsA, tid);
      if (k0 + 32 < K1)
        __builtin_prefetch(W1p + (((size_t)((k0 + 32) >> 5) * (NG >> 4) + ntile) << 9), 0, 1);
      __syncthreads();
      v16bf bf = load_b_packed(W1p, NG >> 4, k0, ntile, lane);
#pragma unroll
      for (int i = 0; i < 4; ++i) {
        v16bf af = load_a_lds(ldsA, (rb0 + i) * 16, lane);
        acc[i] = __builtin_amdgcn_wmma_f32_16x16x32_bf16(false, af, false, bf,
                                                         (short)0, acc[i], false, false);
      }
      __syncthreads();
    }
  }
  // Part 2: recurrent input hprev (f32, K = H)
  for (int k0 = 0; k0 < Hh; k0 += 32) {
    stage_f32_panel(hprev, (long)Hh, k0, ldsA, tid);
    if (k0 + 32 < Hh)
      __builtin_prefetch(W2p + (((size_t)((k0 + 32) >> 5) * (NG >> 4) + ntile) << 9), 0, 1);
    __syncthreads();
    v16bf bf = load_b_packed(W2p, NG >> 4, k0, ntile, lane);
#pragma unroll
    for (int i = 0; i < 4; ++i) {
      v16bf af = load_a_lds(ldsA, (rb0 + i) * 16, lane);
      acc[i] = __builtin_amdgcn_wmma_f32_16x16x32_bf16(false, af, false, bf,
                                                       (short)0, acc[i], false, false);
    }
    __syncthreads();
  }
  // Stage gates: C/D layout -> m = rowblk*16 + v + half*8, n = lane&15
  {
    int half = lane >> 4, r = lane & 15;
#pragma unroll
    for (int i = 0; i < 4; ++i) {
      int mb = (rb0 + i) * 16 + half * 8;
#pragma unroll
      for (int v = 0; v < 8; ++v) ldsG[gate][mb + v][r] = acc[i][v];
    }
  }
  __syncthreads();
  // Fused activations / state update (8 fixed iterations: 128x16 elems / 256 thr)
#pragma unroll
  for (int ii = 0; ii < 8; ++ii) {
    int e = tid + ii * 256;
    int b = e >> 4, n = e & 15;
    int j = jblk * 16 + n;
    float gi = ldsG[0][b][n] + bias[0 * Hh + j];
    float gf = ldsG[1][b][n] + bias[1 * Hh + j];
    float gg = ldsG[2][b][n] + bias[2 * Hh + j];
    float go = ldsG[3][b][n] + bias[3 * Hh + j];
    if constexpr (MODE == 0) {  // encoder layer 0: x_t @ Wih with K=2, done scalar
      float x0 = x2[(long)b * x2_ld + 0];
      float x1 = x2[(long)b * x2_ld + 1];
      gi += x0 * Wih2[0 * NG + 0 * Hh + j] + x1 * Wih2[1 * NG + 0 * Hh + j];
      gf += x0 * Wih2[0 * NG + 1 * Hh + j] + x1 * Wih2[1 * NG + 1 * Hh + j];
      gg += x0 * Wih2[0 * NG + 2 * Hh + j] + x1 * Wih2[1 * NG + 2 * Hh + j];
      go += x0 * Wih2[0 * NG + 3 * Hh + j] + x1 * Wih2[1 * NG + 3 * Hh + j];
    }
    long o = (long)b * Hh + j;
    float cn = sigm(gf) * c_io[o] + sigm(gi) * tanhf(gg);
    float hn = sigm(go) * tanhf(cn);
    c_io[o] = cn;
    h_next[o] = hn;
    if (yout) yout[(long)b * yout_ld + yout_off + j] = f2bfu(hn);
  }
}

// ---------------- generic M=128 GEMM + bias (WMMA) ----------------
__global__ __launch_bounds__(256) void gemm_bias_kernel(
    const float* A, long lda, int K,
    const unsigned short* Bp, int NT,
    const float* bias, float* C, long ldc)
{
  __shared__ unsigned short ldsA[Bq * 32];
  const int tid = threadIdx.x;
  const int wave = tid >> 5, lane = tid & 31;
  const int nt = blockIdx.x;
  v8f acc = {};
  for (int k0 = 0; k0 < K; k0 += 32) {
    stage_f32_panel(A, lda, k0, ldsA, tid);
    if (k0 + 32 < K)
      __builtin_prefetch(Bp + (((size_t)((k0 + 32) >> 5) * NT + nt) << 9), 0, 1);
    __syncthreads();
    v16bf bf = load_b_packed(Bp, NT, k0, nt, lane);
    v16bf af = load_a_lds(ldsA, wave * 16, lane);
    acc = __builtin_amdgcn_wmma_f32_16x16x32_bf16(false, af, false, bf,
                                                  (short)0, acc, false, false);
    __syncthreads();
  }
  int half = lane >> 4, r = lane & 15;
  int n = nt * 16 + r;
  float bv = bias ? bias[n] : 0.0f;
#pragma unroll
  for (int v = 0; v < 8; ++v) {
    int m = wave * 16 + half * 8 + v;
    C[(long)m * ldc + n] = acc[v] + bv;
  }
}

// ---------------- attention pieces (tiny; VALU) ----------------
__global__ __launch_bounds__(256) void attn_scores_kernel(const unsigned short* enc,
                                                          const float* query, float* scores) {
  int i = blockIdx.x * 256 + threadIdx.x;      // over B*T
  if (i >= Bq * Tt) return;
  int b = i >> 8;
  const unsigned short* e = enc + (long)i * (2 * Hh);
  const float* q = query + (long)b * (2 * Hh);
  float s = 0.0f;
  for (int k = 0; k < 2 * Hh; ++k) s += bf2f(e[k]) * q[k];
  scores[i] = s;
}

__global__ __launch_bounds__(256) void softmax_kernel(const float* scores, float* attn,
                                                      float* att_out, int step) {
  __shared__ float red[Tt];
  int b = blockIdx.x, t = threadIdx.x;
  float v = scores[b * Tt + t];
  red[t] = v; __syncthreads();
  for (int s = Tt / 2; s > 0; s >>= 1) { if (t < s) red[t] = fmaxf(red[t], red[t + s]); __syncthreads(); }
  float m = red[0]; __syncthreads();
  float e = __expf(v - m);
  red[t] = e; __syncthreads();
  for (int s = Tt / 2; s > 0; s >>= 1) { if (t < s) red[t] += red[t + s]; __syncthreads(); }
  float a = e / red[0];
  attn[b * Tt + t] = a;
  att_out[((long)b * Tt + t) * LWn + step] = a;   // attention[B,T,LW]
}

__global__ __launch_bounds__(256) void context_kernel(const float* attn,
                                                      const unsigned short* enc, float* catb) {
  long i = (long)blockIdx.x * 256 + threadIdx.x;  // over B * 2H
  if (i >= (long)Bq * 2 * Hh) return;
  int b = (int)(i >> 10), e = (int)(i & (2 * Hh - 1));
  const float* a = attn + b * Tt;
  const unsigned short* ep = enc + ((long)b * Tt) * (2 * Hh) + e;
  float s = 0.0f;
  for (int t = 0; t < Tt; ++t) s += a[t] * bf2f(ep[(long)t * (2 * Hh)]);
  catb[(long)b * (3 * Hh) + Hh + e] = s;          // context part of concat
}

__global__ __launch_bounds__(256) void copy_decout_kernel(const float* h1, float* catb) {
  int i = blockIdx.x * 256 + threadIdx.x;
  if (i >= Bq * Hh) return;
  int b = i >> 9, j = i & (Hh - 1);
  catb[(long)b * (3 * Hh) + j] = h1[i];
}

__global__ void argmax_kernel(const float* logits, long ldr, int* tok) {
  int b = threadIdx.x;
  if (b >= Bq) return;
  const float* p = logits + (long)b * ldr;
  float best = p[0]; int bi = 0;
  for (int v = 1; v < Vv; ++v) { float x = p[v]; if (x > best) { best = x; bi = v; } }
  tok[b] = bi;
}

// ---------------- host orchestration ----------------
extern "C" void kernel_launch(void* const* d_in, const int* in_sizes, int n_in,
                              void* d_out, int out_size, void* d_ws, size_t ws_size,
                              hipStream_t stream) {
  (void)in_sizes; (void)n_in; (void)out_size; (void)ws_size;
  const float* x     = (const float*)d_in[0];
  const float* e0Wih = (const float*)d_in[1];
  const float* e0Whh = (const float*)d_in[2];
  const float* e0b   = (const float*)d_in[3];
  const float* e1Wih = (const float*)d_in[4];
  const float* e1Whh = (const float*)d_in[5];
  const float* e1b   = (const float*)d_in[6];
  const float* dWih  = (const float*)d_in[7];
  const float* dWhh  = (const float*)d_in[8];
  const float* db    = (const float*)d_in[9];
  const float* emb   = (const float*)d_in[10];
  const float* Wq    = (const float*)d_in[11];
  const float* bq    = (const float*)d_in[12];
  const float* Wc    = (const float*)d_in[13];
  const float* bc    = (const float*)d_in[14];
  const float* Wf    = (const float*)d_in[15];
  const float* bfv   = (const float*)d_in[16];

  float* out        = (float*)d_out;
  float* out_logits = out;                               // [B, LW, V]
  float* out_hF     = out + (long)Bq * LWn * Vv;         // +524288
  float* out_cF     = out_hF + (long)2 * Bq * Hh;        // +131072
  float* out_att    = out_cF + (long)2 * Bq * Hh;        // [B, T, LW]

  char* wp = (char*)d_ws;
  auto alloc = [&](size_t bytes) -> void* {
    void* p = (void*)wp; wp += (bytes + 255) & ~(size_t)255; return p;
  };

  unsigned short *pk_e0Whh[2], *pk_e1Wih[2], *pk_e1Whh[2], *pk_dWih[2], *pk_dWhh[2];
  for (int d = 0; d < 2; ++d) pk_e0Whh[d] = (unsigned short*)alloc((size_t)512 * 2048 * 2);
  for (int d = 0; d < 2; ++d) pk_e1Wih[d] = (unsigned short*)alloc((size_t)1024 * 2048 * 2);
  for (int d = 0; d < 2; ++d) pk_e1Whh[d] = (unsigned short*)alloc((size_t)512 * 2048 * 2);
  for (int l = 0; l < 2; ++l) pk_dWih[l]  = (unsigned short*)alloc((size_t)512 * 2048 * 2);
  for (int l = 0; l < 2; ++l) pk_dWhh[l]  = (unsigned short*)alloc((size_t)512 * 2048 * 2);
  unsigned short* pk_Wq = (unsigned short*)alloc((size_t)512 * 1024 * 2);
  unsigned short* pk_Wc = (unsigned short*)alloc((size_t)1536 * 512 * 2);
  unsigned short* pk_Wf = (unsigned short*)alloc((size_t)512 * 128 * 2);
  unsigned short* emb_bf = (unsigned short*)alloc((size_t)Vv * Hh * 2);
  unsigned short* y0  = (unsigned short*)alloc((size_t)Bq * Tt * 1024 * 2);  // layer0 bidir out (bf16)
  unsigned short* enc = (unsigned short*)alloc((size_t)Bq * Tt * 1024 * 2);  // enc_outs (bf16)
  float* hA[2];  hA[0]  = (float*)alloc(65536 * 4); hA[1]  = (float*)alloc(65536 * 4);
  float* hB2[2]; hB2[0] = (float*)alloc(65536 * 4); hB2[1] = (float*)alloc(65536 * 4);
  float* cA  = (float*)alloc(65536 * 4);
  float* cB2 = (float*)alloc(65536 * 4);
  float* dec_h[2]; dec_h[0] = (float*)alloc(2 * 65536 * 4); dec_h[1] = (float*)alloc(2 * 65536 * 4);
  float* dec_c = (float*)alloc(2 * 65536 * 4);
  float* query = (float*)alloc(131072 * 4);
  float* scores = (float*)alloc(32768 * 4);
  float* attn   = (float*)alloc(32768 * 4);
  float* catb   = (float*)alloc(196608 * 4);
  float* comb   = (float*)alloc(65536 * 4);
  int* tok = (int*)alloc(128 * 4);

  auto packW = [&](const float* src, int K, int N, unsigned short* dst) {
    long n = (long)K * N;
    pack_b_kernel<<<(unsigned)((n + 255) / 256), 256, 0, stream>>>(src, K, N, dst);
  };
  auto zerof = [&](float* p, long n) {
    zero_kernel<<<(unsigned)((n + 255) / 256), 256, 0, stream>>>(p, n);
  };

  // Weight pre-pack (bf16 fragment-linear; done each call — deterministic)
  for (int d = 0; d < 2; ++d) packW(e0Whh + (size_t)d * 512 * 2048, 512, 2048, pk_e0Whh[d]);
  for (int d = 0; d < 2; ++d) packW(e1Wih + (size_t)d * 1024 * 2048, 1024, 2048, pk_e1Wih[d]);
  for (int d = 0; d < 2; ++d) packW(e1Whh + (size_t)d * 512 * 2048, 512, 2048, pk_e1Whh[d]);
  for (int l = 0; l < 2; ++l) packW(dWih + (size_t)l * 512 * 2048, 512, 2048, pk_dWih[l]);
  for (int l = 0; l < 2; ++l) packW(dWhh + (size_t)l * 512 * 2048, 512, 2048, pk_dWhh[l]);
  packW(Wq, 512, 1024, pk_Wq);
  packW(Wc, 1536, 512, pk_Wc);
  packW(Wf, 512, 128, pk_Wf);
  cvt_bf_kernel<<<256, 256, 0, stream>>>(emb, emb_bf, (long)Vv * Hh);

  // ---------- encoder layer 0 (bidirectional) ----------
  zerof(hA[0], 65536); zerof(hA[1], 65536); zerof(cA, 65536);
  zerof(hB2[0], 65536); zerof(hB2[1], 65536); zerof(cB2, 65536);
  for (int s = 0; s < Tt; ++s) {
    int tf = s, tb = Tt - 1 - s;
    lstm_step_kernel<0><<<32, 256, 0, stream>>>(
        nullptr, nullptr, nullptr, 0L, 0, nullptr,
        hA[s & 1], pk_e0Whh[0], e0b,
        x + (size_t)tf * 2, (long)(Tt * 2), e0Wih,
        cA, hA[(s + 1) & 1],
        y0 + (size_t)tf * 1024, (long)(Tt * 1024), 0L);
    lstm_step_kernel<0><<<32, 256, 0, stream>>>(
        nullptr, nullptr, nullptr, 0L, 0, nullptr,
        hB2[s & 1], pk_e0Whh[1], e0b + 2048,
        x + (size_t)tb * 2, (long)(Tt * 2), e0Wih + 2 * 2048,
        cB2, hB2[(s + 1) & 1],
        y0 + (size_t)tb * 1024, (long)(Tt * 1024), 512L);
  }
  add_kernel<<<256, 256, 0, stream>>>(hA[0], hB2[0], dec_h[0] + 0, 65536);
  add_kernel<<<256, 256, 0, stream>>>(cA, cB2, dec_c + 0, 65536);

  // ---------- encoder layer 1 (bidirectional, input = y0 bf16) ----------
  zerof(hA[0], 65536); zerof(hA[1], 65536); zerof(cA, 65536);
  zerof(hB2[0], 65536); zerof(hB2[1], 65536); zerof(cB2, 65536);
  for (int s = 0; s < Tt; ++s) {
    int tf = s, tb = Tt - 1 - s;
    lstm_step_kernel<1><<<32, 256, 0, stream>>>(
        y0 + (size_t)tf * 1024, nullptr, nullptr, (long)(Tt * 1024), 1024, pk_e1Wih[0],
        hA[s & 1], pk_e1Whh[0], e1b,
        nullptr, 0L, nullptr,
        cA, hA[(s + 1) & 1],
        enc + (size_t)tf * 1024, (long)(Tt * 1024), 0L);
    lstm_step_kernel<1><<<32, 256, 0, stream>>>(
        y0 + (size_t)tb * 1024, nullptr, nullptr, (long)(Tt * 1024), 1024, pk_e1Wih[1],
        hB2[s & 1], pk_e1Whh[1], e1b + 2048,
        nullptr, 0L, nullptr,
        cB2, hB2[(s + 1) & 1],
        enc + (size_t)tb * 1024, (long)(Tt * 1024), 512L);
  }
  add_kernel<<<256, 256, 0, stream>>>(hA[0], hB2[0], dec_h[0] + 65536, 65536);
  add_kernel<<<256, 256, 0, stream>>>(cA, cB2, dec_c + 65536, 65536);

  // ---------- greedy attention decoder ----------
  set_tok_kernel<<<1, 128, 0, stream>>>(tok);
  for (int st = 0; st < LWn; ++st) {
    int p = st & 1;
    // layer 0: input = emb[tok]
    lstm_step_kernel<2><<<32, 256, 0, stream>>>(
        emb_bf, nullptr, tok, (long)Hh, Hh, pk_dWih[0],
        dec_h[p] + 0, pk_dWhh[0], db,
        nullptr, 0L, nullptr,
        dec_c + 0, dec_h[1 - p] + 0,
        nullptr, 0L, 0L);
    // layer 1: input = new h0 (f32)
    lstm_step_kernel<3><<<32, 256, 0, stream>>>(
        nullptr, dec_h[1 - p] + 0, nullptr, (long)Hh, Hh, pk_dWih[1],
        dec_h[p] + 65536, pk_dWhh[1], db + 2048,
        nullptr, 0L, nullptr,
        dec_c + 65536, dec_h[1 - p] + 65536,
        nullptr, 0L, 0L);
    const float* dec_out = dec_h[1 - p] + 65536;
    copy_decout_kernel<<<256, 256, 0, stream>>>(dec_out, catb);
    // query = dec_out @ Wq + bq   [128,1024]
    gemm_bias_kernel<<<64, 256, 0, stream>>>(dec_out, (long)Hh, Hh, pk_Wq, 64, bq, query, 1024L);
    attn_scores_kernel<<<128, 256, 0, stream>>>(enc, query, scores);
    softmax_kernel<<<128, 256, 0, stream>>>(scores, attn, out_att, st);
    context_kernel<<<512, 256, 0, stream>>>(attn, enc, catb);
    // combined = [dec_out, context] @ Wc + bc   [128,512]
    gemm_bias_kernel<<<32, 256, 0, stream>>>(catb, 1536L, 1536, pk_Wc, 32, bc, comb, 512L);
    // logits -> d_out[b, st, :]
    gemm_bias_kernel<<<8, 256, 0, stream>>>(comb, (long)Hh, Hh, pk_Wf, 8, bfv,
                                            out_logits + (size_t)st * Vv, (long)(LWn * Vv));
    argmax_kernel<<<1, 128, 0, stream>>>(out_logits + (size_t)st * Vv, (long)(LWn * Vv), tok);
  }
  // final decoder states (LW=32 even -> parity buffer 0)
  copyf_kernel<<<512, 256, 0, stream>>>(dec_h[0], out_hF, 131072);
  copyf_kernel<<<512, 256, 0, stream>>>(dec_c, out_cF, 131072);
}
